// Decoder_20023137535000
// MI455X (gfx1250) — matrix operand order repaired and checked by reading, not compiled, not run on hardware
//
#include <hip/hip_runtime.h>

// ---------------------------------------------------------------------------
// Transformer decoder for gfx1250 (MI455X).
// GEMMs: v_wmma_f32_16x16x32_f16 with double-buffered async-DMA LDS staging
// (global_load_async_to_lds_b128 + s_wait_asynccnt, ds_load_b128 fragments).
// ---------------------------------------------------------------------------

typedef __attribute__((ext_vector_type(16))) _Float16 v16h;
typedef __attribute__((ext_vector_type(8)))  _Float16 v8h;
typedef __attribute__((ext_vector_type(8)))  float    v8f;

static constexpr int Dm  = 512;    // d_model
static constexpr int Hh  = 8;      // heads
static constexpr int DK  = 64;     // head dim
static constexpr int FFd = 2048;   // feedforward
static constexpr int Ll  = 6;      // layers
static constexpr int Bb  = 16;     // batch
static constexpr int Ss  = 512;    // seq len (dec == enc)
static constexpr int MR  = Bb * Ss; // 8192 rows of activations
static constexpr int PADK = 40;    // LDS row pitch in halves (80B): conflict-free b128 reads

enum { EPI_PLAIN = 0, EPI_SCATTER_QK = 1, EPI_SCATTER_VT = 2, EPI_GATHER_CTX = 3 };

__device__ __forceinline__ void async_b128_to_lds(unsigned lds_off, const void* gaddr)
{
    // CDNA5 async DMA: LDS[lds_off] <= MEM[gaddr], 16B, tracked by ASYNCcnt.
    asm volatile("global_load_async_to_lds_b128 %0, %1, off"
                 :: "v"(lds_off), "v"((unsigned long long)(size_t)gaddr)
                 : "memory");
}
__device__ __forceinline__ void wait_async0()
{
    asm volatile("s_wait_asynccnt 0" ::: "memory");
}

// ---------------------------------------------------------------------------
// WMMA GEMM:  C[M,N] = A[M,K] (f16, row-major) x B[K,N]  with B supplied
// TRANSPOSED as Bt[N,K] (f16, row-major).  8 waves / block (NWM x NWN), each
// wave computes WM x WN 16x16 tiles.  K is consumed in 32-wide steps; each
// step's A (BMx32) and Bt (BNx32) tiles are staged in LDS by async DMA with
// double buffering so the DMA of step k+1 overlaps the WMMAs of step k.
// Batched over blockIdx.z with strides sA/sB/sC.
// ---------------------------------------------------------------------------
template <int NWM, int NWN, int WM, int WN, int MODE, bool RELU, typename OutT>
__global__ __launch_bounds__(NWM * NWN * 32)
void gemm_wmma(const _Float16* __restrict__ A, const _Float16* __restrict__ Bt,
               const float* __restrict__ bias, OutT* __restrict__ C,
               int M, int N, int K, long sA, long sB, long sC)
{
    constexpr int NT = NWM * NWN * 32;
    constexpr int BM = NWM * WM * 16;
    constexpr int BN = NWN * WN * 16;

    __shared__ _Float16 shA[2][BM * PADK];
    __shared__ _Float16 shB[2][BN * PADK];

    const int tid  = threadIdx.x;
    const int lane = tid & 31;
    const int wave = tid >> 5;
    const int wm   = wave / NWN;
    const int wn   = wave % NWN;
    const int z    = blockIdx.z;

    A  += (long)z * sA;
    Bt += (long)z * sB;

    const int mBlk = blockIdx.y * BM;
    const int nBlk = blockIdx.x * BN;

    const int rsel = lane & 15;   // row (A) / col (B) within a 16-tile
    const int ksel = lane >> 4;   // which K-half this lane holds

    // cooperative async stage of one (BMx32, BNx32) K-slice into buffer p
    auto stage = [&](int p, int kk) {
#pragma unroll
        for (int i = tid; i < BM * 4; i += NT) {
            const int row = i >> 2, ch = i & 3;            // 4 x 16B chunks / row
            async_b128_to_lds((unsigned)(size_t)&shA[p][row * PADK + ch * 8],
                              A + (long)(mBlk + row) * K + kk + ch * 8);
        }
#pragma unroll
        for (int i = tid; i < BN * 4; i += NT) {
            const int row = i >> 2, ch = i & 3;
            async_b128_to_lds((unsigned)(size_t)&shB[p][row * PADK + ch * 8],
                              Bt + (long)(nBlk + row) * K + kk + ch * 8);
        }
    };

    v8f acc[WM][WN];
    const v8f vzero = {0.f, 0.f, 0.f, 0.f, 0.f, 0.f, 0.f, 0.f};
#pragma unroll
    for (int i = 0; i < WM; ++i)
#pragma unroll
        for (int j = 0; j < WN; ++j) acc[i][j] = vzero;

    stage(0, 0);
    wait_async0();
    __syncthreads();

    for (int kk = 0; kk < K; kk += 32) {
        const int  p    = (kk >> 5) & 1;
        const bool more = (kk + 32) < K;
        if (more) stage(1 - p, kk + 32);   // DMA next slice while we compute

        v16h af[WM], bf[WN];
#pragma unroll
        for (int i = 0; i < WM; ++i) {
            // A 16x32 layout: lanes 0-15 hold K 0..7 & 16..23, lanes 16-31 K 8..15 & 24..31
            const _Float16* pa = &shA[p][(wm * WM * 16 + i * 16 + rsel) * PADK + ksel * 8];
            v8h lo = *(const v8h*)pa;
            v8h hi = *(const v8h*)(pa + 16);
            af[i] = __builtin_shufflevector(lo, hi, 0, 1, 2, 3, 4, 5, 6, 7,
                                            8, 9, 10, 11, 12, 13, 14, 15);
        }
#pragma unroll
        for (int j = 0; j < WN; ++j) {
            // B 32x16 layout: lanes 0-15 hold K 0..7 & 16..23, lanes 16-31 K 8..15 & 24..31
            const _Float16* pb = &shB[p][(wn * WN * 16 + j * 16 + rsel) * PADK + ksel * 8];
            v8h lo = *(const v8h*)pb;
            v8h hi = *(const v8h*)(pb + 16);
            bf[j] = __builtin_shufflevector(lo, hi, 0, 1, 2, 3, 4, 5, 6, 7,
                                            8, 9, 10, 11, 12, 13, 14, 15);
        }
#pragma unroll
        for (int i = 0; i < WM; ++i)
#pragma unroll
            for (int j = 0; j < WN; ++j)
                acc[i][j] = __builtin_amdgcn_wmma_f32_16x16x32_f16(
                    false, af[i], false, bf[j], (short)0, acc[i][j], false, false);

        if (more) {                 // own DMAs done, then block-wide visibility
            wait_async0();
            __syncthreads();
        }
    }

    // Epilogue.  C/D layout: VGPR r -> M = r + (lane>=16 ? 8 : 0), N = lane&15.
    const int cn     = lane & 15;
    const int rowoff = (lane >> 4) * 8;
#pragma unroll
    for (int i = 0; i < WM; ++i)
#pragma unroll
        for (int j = 0; j < WN; ++j) {
            const int tn = nBlk + wn * WN * 16 + j * 16 + cn;
            float bv = 0.f;
            if (MODE != EPI_GATHER_CTX && bias != nullptr) bv = bias[tn];
#pragma unroll
            for (int r = 0; r < 8; ++r) {
                const int m = mBlk + wm * WM * 16 + i * 16 + rowoff + r;
                float val = acc[i][j][r] + bv;
                if (RELU) val = fmaxf(val, 0.f);
                long idx;
                if (MODE == EPI_PLAIN) {
                    idx = (long)z * sC + (long)m * N + tn;
                } else if (MODE == EPI_SCATTER_QK) {
                    const int b = m / Ss, s = m % Ss, h = tn / DK, d = tn % DK;
                    idx = (((long)b * Hh + h) * Ss + s) * DK + d;   // [B,H,S,dk]
                } else if (MODE == EPI_SCATTER_VT) {
                    const int b = m / Ss, s = m % Ss, h = tn / DK, d = tn % DK;
                    idx = (((long)b * Hh + h) * DK + d) * Ss + s;   // [B,H,dk,S]
                } else { // EPI_GATHER_CTX: z = b*H+h, m = s, tn = d
                    const int b = z / Hh, h = z % Hh;
                    idx = ((long)b * Ss + m) * Dm + h * DK + tn;    // [B,S,D]
                }
                C[idx] = (OutT)val;
            }
        }
}

// ---------------------------------------------------------------------------
// Elementwise / reduction kernels
// ---------------------------------------------------------------------------

__global__ __launch_bounds__(256)
void embed_pe_kernel(const int* __restrict__ ids, const float* __restrict__ emb,
                     const float* __restrict__ pe, _Float16* __restrict__ out16)
{
    const int row = blockIdx.x;           // b*Ss + s
    const int s   = row % Ss;
    const int tok = ids[row];
    const float* e = emb + (long)tok * Dm;
    const float* p = pe + (long)s * Dm;
    for (int d = threadIdx.x; d < Dm; d += 256)
        out16[(long)row * Dm + d] = (_Float16)(e[d] + p[d]);
}

__global__ __launch_bounds__(256)
void cvt_f32_to_f16(const float* __restrict__ in, _Float16* __restrict__ out, long n)
{
    long i = (long)blockIdx.x * 256 + threadIdx.x;
    if (i < n) out[i] = (_Float16)in[i];
}

// W[K,N] f32  ->  Wt[N,K] f16 (tiled transpose, coalesced both sides)
__global__ __launch_bounds__(256)
void transpose_cvt(const float* __restrict__ W, _Float16* __restrict__ Wt, int K, int N)
{
    __shared__ float tile[32][33];
    const int k0 = blockIdx.y * 32, n0 = blockIdx.x * 32;
    const int tx = threadIdx.x, ty = threadIdx.y;  // (32, 8)
    for (int r = ty; r < 32; r += 8) tile[r][tx] = W[(long)(k0 + r) * N + n0 + tx];
    __syncthreads();
    for (int r = ty; r < 32; r += 8)
        Wt[(long)(n0 + r) * K + k0 + tx] = (_Float16)tile[tx][r];
}

// masked softmax over a 512-wide row; one 256-thread block per (b,h,q) row
template <bool CAUSAL>
__global__ __launch_bounds__(256)
void softmax_mask(const float* __restrict__ sc, _Float16* __restrict__ at,
                  const int* __restrict__ key_ids, const int* __restrict__ pad_ptr,
                  float scale)
{
    const int row = blockIdx.x;              // (b*H + h)*Sq + q
    const int q   = row % Ss;
    const int b   = row / (Hh * Ss);
    const int tid = threadIdx.x;
    const int pad = pad_ptr[0];
    const float* src = sc + (long)row * Ss;

    float v[2];
#pragma unroll
    for (int t = 0; t < 2; ++t) {
        const int k = tid + t * 256;
        const bool msk = (key_ids[b * Ss + k] == pad) || (CAUSAL && (k > q));
        v[t] = msk ? -1e9f : src[k] * scale;
    }

    __shared__ float red[256];
    red[tid] = fmaxf(v[0], v[1]);
    __syncthreads();
    for (int st = 128; st > 0; st >>= 1) {
        if (tid < st) red[tid] = fmaxf(red[tid], red[tid + st]);
        __syncthreads();
    }
    const float mx = red[0];
    __syncthreads();

    const float e0 = __expf(v[0] - mx), e1 = __expf(v[1] - mx);
    red[tid] = e0 + e1;
    __syncthreads();
    for (int st = 128; st > 0; st >>= 1) {
        if (tid < st) red[tid] += red[tid + st];
        __syncthreads();
    }
    const float inv = 1.f / red[0];

    _Float16* dst = at + (long)row * Ss;
    dst[tid]       = (_Float16)(e0 * inv);
    dst[tid + 256] = (_Float16)(e1 * inv);
}

// out = LayerNorm(y + res) * g + beta  ->  f32 master + f16 copy
__global__ __launch_bounds__(256)
void add_ln_kernel(const float* __restrict__ y, const float* __restrict__ res,
                   const float* __restrict__ g, const float* __restrict__ bt,
                   float* __restrict__ o32, _Float16* __restrict__ o16)
{
    const int row = blockIdx.x;
    const int tid = threadIdx.x;
    const long base = (long)row * Dm;
    float v0 = y[base + tid]       + res[base + tid];
    float v1 = y[base + tid + 256] + res[base + tid + 256];

    __shared__ float red[256];
    red[tid] = v0 + v1;
    __syncthreads();
    for (int st = 128; st > 0; st >>= 1) {
        if (tid < st) red[tid] += red[tid + st];
        __syncthreads();
    }
    const float mean = red[0] * (1.f / Dm);
    __syncthreads();

    const float c0 = v0 - mean, c1 = v1 - mean;
    red[tid] = c0 * c0 + c1 * c1;
    __syncthreads();
    for (int st = 128; st > 0; st >>= 1) {
        if (tid < st) red[tid] += red[tid + st];
        __syncthreads();
    }
    const float rstd = rsqrtf(red[0] * (1.f / Dm) + 1e-5f);

    const float r0 = c0 * rstd * g[tid]       + bt[tid];
    const float r1 = c1 * rstd * g[tid + 256] + bt[tid + 256];
    o32[base + tid]       = r0;
    o32[base + tid + 256] = r1;
    o16[base + tid]       = (_Float16)r0;
    o16[base + tid + 256] = (_Float16)r1;
}

// ---------------------------------------------------------------------------
// Host-side launch helpers
// ---------------------------------------------------------------------------
template <int NWM, int NWN, int WM, int WN, int MODE, bool RELU, typename OutT>
static void gemm(hipStream_t st, const _Float16* A, const _Float16* Bt,
                 const float* bias, OutT* C, int M, int N, int K,
                 int batch, long sA, long sB, long sC)
{
    constexpr int BM = NWM * WM * 16, BN = NWN * WN * 16;
    dim3 grid(N / BN, M / BM, batch), block(NWM * NWN * 32);
    gemm_wmma<NWM, NWN, WM, WN, MODE, RELU, OutT>
        <<<grid, block, 0, st>>>(A, Bt, bias, C, M, N, K, sA, sB, sC);
}

static void tconv(hipStream_t st, const float* W, _Float16* Wt, int K, int N)
{
    dim3 grid(N / 32, K / 32), block(32, 8);
    transpose_cvt<<<grid, block, 0, st>>>(W, Wt, K, N);
}

extern "C" void kernel_launch(void* const* d_in, const int* in_sizes, int n_in,
                              void* d_out, int out_size, void* d_ws, size_t ws_size,
                              hipStream_t stream)
{
    (void)in_sizes; (void)n_in; (void)out_size; (void)ws_size;

    const int*   dec_inputs = (const int*)d_in[0];
    const int*   enc_inputs = (const int*)d_in[1];
    const float* enc_out    = (const float*)d_in[2];
    const int*   pad_ptr    = (const int*)d_in[3];
    const float* emb        = (const float*)d_in[4];
    const float* pe         = (const float*)d_in[5];
    const float* sub_w1     = (const float*)d_in[6];
    const float* sub_b1     = (const float*)d_in[7];
    const float* sub_w2     = (const float*)d_in[8];
    const float* sub_b2     = (const float*)d_in[9];
    const float* sa_wq = (const float*)d_in[10], *sa_bq = (const float*)d_in[11];
    const float* sa_wk = (const float*)d_in[12], *sa_bk = (const float*)d_in[13];
    const float* sa_wv = (const float*)d_in[14], *sa_bv = (const float*)d_in[15];
    const float* sa_wo = (const float*)d_in[16], *sa_bo = (const float*)d_in[17];
    const float* ca_wq = (const float*)d_in[18], *ca_bq = (const float*)d_in[19];
    const float* ca_wk = (const float*)d_in[20], *ca_bk = (const float*)d_in[21];
    const float* ca_wv = (const float*)d_in[22], *ca_bv = (const float*)d_in[23];
    const float* ca_wo = (const float*)d_in[24], *ca_bo = (const float*)d_in[25];
    const float* sa_g  = (const float*)d_in[26], *sa_beta = (const float*)d_in[27];
    const float* ca_g  = (const float*)d_in[28], *ca_beta = (const float*)d_in[29];
    const float* ff_w1 = (const float*)d_in[30], *ff_b1 = (const float*)d_in[31];
    const float* ff_w2 = (const float*)d_in[32], *ff_b2 = (const float*)d_in[33];
    const float* ff_g  = (const float*)d_in[34], *ff_beta = (const float*)d_in[35];
    float* out = (float*)d_out;

    // ---- workspace bump allocator (~356 MiB total) ----
    char*  base = (char*)d_ws;
    size_t off  = 0;
    auto alloc = [&](size_t bytes) -> char* {
        off = (off + 255) & ~(size_t)255;
        char* p = base + off;
        off += bytes;
        return p;
    };
    _Float16* x16    = (_Float16*)alloc((size_t)MR * Dm * 2);
    float*    x32    = (float*)   alloc((size_t)MR * Dm * 4);
    float*    y32    = (float*)   alloc((size_t)MR * Dm * 4);
    _Float16* h16    = (_Float16*)alloc((size_t)MR * FFd * 2);
    _Float16* q16    = (_Float16*)alloc((size_t)Bb * Hh * Ss * DK * 2);
    _Float16* k16    = (_Float16*)alloc((size_t)Bb * Hh * Ss * DK * 2);
    _Float16* vT16   = (_Float16*)alloc((size_t)Bb * Hh * DK * Ss * 2);
    _Float16* ctx16  = (_Float16*)alloc((size_t)MR * Dm * 2);
    float*    sc32   = (float*)   alloc((size_t)Bb * Hh * Ss * Ss * 4);
    _Float16* attn16 = (_Float16*)alloc((size_t)Bb * Hh * Ss * Ss * 2);
    _Float16* enc16  = (_Float16*)alloc((size_t)MR * Dm * 2);
    _Float16* t_sub1 = (_Float16*)alloc((size_t)FFd * Dm * 2);
    _Float16* t_sub2 = (_Float16*)alloc((size_t)Dm * FFd * 2);
    _Float16* t_sawq = (_Float16*)alloc((size_t)Ll * Dm * Dm * 2);
    _Float16* t_sawk = (_Float16*)alloc((size_t)Ll * Dm * Dm * 2);
    _Float16* t_sawv = (_Float16*)alloc((size_t)Ll * Dm * Dm * 2);
    _Float16* t_sawo = (_Float16*)alloc((size_t)Ll * Dm * Dm * 2);
    _Float16* t_cawq = (_Float16*)alloc((size_t)Ll * Dm * Dm * 2);
    _Float16* t_cawk = (_Float16*)alloc((size_t)Ll * Dm * Dm * 2);
    _Float16* t_cawv = (_Float16*)alloc((size_t)Ll * Dm * Dm * 2);
    _Float16* t_cawo = (_Float16*)alloc((size_t)Ll * Dm * Dm * 2);
    _Float16* t_ff1  = (_Float16*)alloc((size_t)Ll * FFd * Dm * 2);
    _Float16* t_ff2  = (_Float16*)alloc((size_t)Ll * Dm * FFd * 2);

    // ---- one-time (per call) weight conversion: f32 [K,N] -> f16 [N,K] ----
    tconv(stream, sub_w1, t_sub1, Dm, FFd);
    tconv(stream, sub_w2, t_sub2, FFd, Dm);
    for (int l = 0; l < Ll; ++l) {
        const size_t wo = (size_t)l * Dm * Dm;
        tconv(stream, sa_wq + wo, t_sawq + wo, Dm, Dm);
        tconv(stream, sa_wk + wo, t_sawk + wo, Dm, Dm);
        tconv(stream, sa_wv + wo, t_sawv + wo, Dm, Dm);
        tconv(stream, sa_wo + wo, t_sawo + wo, Dm, Dm);
        tconv(stream, ca_wq + wo, t_cawq + wo, Dm, Dm);
        tconv(stream, ca_wk + wo, t_cawk + wo, Dm, Dm);
        tconv(stream, ca_wv + wo, t_cawv + wo, Dm, Dm);
        tconv(stream, ca_wo + wo, t_cawo + wo, Dm, Dm);
        tconv(stream, ff_w1 + (size_t)l * Dm * FFd, t_ff1 + (size_t)l * FFd * Dm, Dm, FFd);
        tconv(stream, ff_w2 + (size_t)l * FFd * Dm, t_ff2 + (size_t)l * Dm * FFd, FFd, Dm);
    }
    cvt_f32_to_f16<<<(MR * Dm) / 256, 256, 0, stream>>>(enc_out, enc16, (long)MR * Dm);

    // ---- embedding + PE, then the two pointwise "subsampling" layers ----
    embed_pe_kernel<<<MR, 256, 0, stream>>>(dec_inputs, emb, pe, x16);
    gemm<2, 4, 2, 2, EPI_PLAIN, true, _Float16>(stream, x16, t_sub1, sub_b1, h16, MR, FFd, Dm, 1, 0, 0, 0);
    gemm<2, 4, 2, 2, EPI_PLAIN, true, float>(stream, h16, t_sub2, sub_b2, x32, MR, Dm, FFd, 1, 0, 0, 0);
    cvt_f32_to_f16<<<(MR * Dm) / 256, 256, 0, stream>>>(x32, x16, (long)MR * Dm);

    const long sQK = (long)Ss * DK;   // per-(b,h) stride of q/k
    const long sVT = (long)DK * Ss;   // per-(b,h) stride of v^T
    const long sSC = (long)Ss * Ss;   // per-(b,h) stride of scores/attn
    const int  ZZ  = Bb * Hh;

    for (int l = 0; l < Ll; ++l) {
        const size_t wo = (size_t)l * Dm * Dm;
        const size_t bo_ = (size_t)l * Dm;

        // ---------------- self-attention ----------------
        gemm<2, 4, 2, 2, EPI_SCATTER_QK, false, _Float16>(stream, x16, t_sawq + wo, sa_bq + bo_, q16,  MR, Dm, Dm, 1, 0, 0, 0);
        gemm<2, 4, 2, 2, EPI_SCATTER_QK, false, _Float16>(stream, x16, t_sawk + wo, sa_bk + bo_, k16,  MR, Dm, Dm, 1, 0, 0, 0);
        gemm<2, 4, 2, 2, EPI_SCATTER_VT, false, _Float16>(stream, x16, t_sawv + wo, sa_bv + bo_, vT16, MR, Dm, Dm, 1, 0, 0, 0);
        // scores = q @ k^T : k in [B,H,S,dk] IS the transposed-B operand
        gemm<2, 4, 2, 2, EPI_PLAIN, false, float>(stream, q16, k16, nullptr, sc32, Ss, Ss, DK, ZZ, sQK, sQK, sSC);
        softmax_mask<true><<<ZZ * Ss, 256, 0, stream>>>(sc32, attn16, dec_inputs, pad_ptr, 0.125f);
        gemm<4, 2, 2, 2, EPI_GATHER_CTX, false, _Float16>(stream, attn16, vT16, nullptr, ctx16, Ss, DK, Ss, ZZ, sSC, sVT, 0);
        gemm<2, 4, 2, 2, EPI_PLAIN, false, float>(stream, ctx16, t_sawo + wo, sa_bo + bo_, y32, MR, Dm, Dm, 1, 0, 0, 0);
        add_ln_kernel<<<MR, 256, 0, stream>>>(y32, x32, sa_g + bo_, sa_beta + bo_, x32, x16);

        // ---------------- cross-attention ----------------
        gemm<2, 4, 2, 2, EPI_SCATTER_QK, false, _Float16>(stream, x16,   t_cawq + wo, ca_bq + bo_, q16,  MR, Dm, Dm, 1, 0, 0, 0);
        gemm<2, 4, 2, 2, EPI_SCATTER_QK, false, _Float16>(stream, enc16, t_cawk + wo, ca_bk + bo_, k16,  MR, Dm, Dm, 1, 0, 0, 0);
        gemm<2, 4, 2, 2, EPI_SCATTER_VT, false, _Float16>(stream, enc16, t_cawv + wo, ca_bv + bo_, vT16, MR, Dm, Dm, 1, 0, 0, 0);
        gemm<2, 4, 2, 2, EPI_PLAIN, false, float>(stream, q16, k16, nullptr, sc32, Ss, Ss, DK, ZZ, sQK, sQK, sSC);
        softmax_mask<false><<<ZZ * Ss, 256, 0, stream>>>(sc32, attn16, enc_inputs, pad_ptr, 0.125f);
        gemm<4, 2, 2, 2, EPI_GATHER_CTX, false, _Float16>(stream, attn16, vT16, nullptr, ctx16, Ss, DK, Ss, ZZ, sSC, sVT, 0);
        gemm<2, 4, 2, 2, EPI_PLAIN, false, float>(stream, ctx16, t_cawo + wo, ca_bo + bo_, y32, MR, Dm, Dm, 1, 0, 0, 0);
        add_ln_kernel<<<MR, 256, 0, stream>>>(y32, x32, ca_g + bo_, ca_beta + bo_, x32, x16);

        // ---------------- feed-forward ----------------
        gemm<2, 4, 2, 2, EPI_PLAIN, true, _Float16>(stream, x16, t_ff1 + (size_t)l * FFd * Dm,
                                                    ff_b1 + (size_t)l * FFd, h16, MR, FFd, Dm, 1, 0, 0, 0);
        gemm<2, 4, 2, 2, EPI_PLAIN, false, float>(stream, h16, t_ff2 + (size_t)l * Dm * FFd,
                                                  ff_b2 + bo_, y32, MR, Dm, FFd, 1, 0, 0, 0);
        float* lnout = (l == Ll - 1) ? out : x32;
        add_ln_kernel<<<MR, 256, 0, stream>>>(y32, x32, ff_g + bo_, ff_beta + bo_, lnout, x16);
    }
}
